// TripletLoss_86758339379639
// MI455X (gfx1250) — matrix operand order repaired and checked by reading, not compiled, not run on hardware
//
#include <hip/hip_runtime.h>
#include <hip/hip_bf16.h>

typedef __attribute__((ext_vector_type(16))) __bf16 v16bf;
typedef __attribute__((ext_vector_type(8)))  float  v8f;

#define BATCH   4096
#define FEAT    256
#define MARGIN  0.3f
#define SLICES  8          // column-dim split for occupancy
#define JT_PER_SLICE 32    // (4096/16)/SLICES column tiles per slice
#define KCHUNKS 8          // FEAT / 32

union FragU { v16bf v; uint4 q[2]; };

__device__ __forceinline__ unsigned short f32_to_bf16_rne(float x) {
    unsigned u = __float_as_uint(x);
    unsigned r = u + 0x7FFFu + ((u >> 16) & 1u);
    return (unsigned short)(r >> 16);
}

// ---------------------------------------------------------------------------
// Kernel 0: split fp32 features into bf16 hi + bf16 lo, compute row sq-norms.
// One 256-thread block per row (FEAT == 256).
// ---------------------------------------------------------------------------
__global__ void __launch_bounds__(256)
triplet_prep_kernel(const float* __restrict__ feats,
                    unsigned short* __restrict__ fhi,
                    unsigned short* __restrict__ flo,
                    float* __restrict__ sqn) {
    const int row = blockIdx.x;
    const int tid = threadIdx.x;
    const int idx = row * FEAT + tid;

    float x = feats[idx];

    unsigned u  = __float_as_uint(x);
    unsigned rh = u + 0x7FFFu + ((u >> 16) & 1u);
    unsigned short h = (unsigned short)(rh >> 16);
    float hf = __uint_as_float((unsigned)h << 16);
    float l  = x - hf;

    fhi[idx] = h;
    flo[idx] = f32_to_bf16_rne(l);

    __shared__ float s[256];
    s[tid] = x * x;
    __syncthreads();
    for (int off = 128; off > 0; off >>= 1) {
        if (tid < off) s[tid] += s[tid + off];
        __syncthreads();
    }
    if (tid == 0) sqn[row] = s[0];
}

// ---------------------------------------------------------------------------
// Kernel 1: fused Gram (bf16 WMMA, hi/lo split-precision) + batch-hard mining.
//   grid = (32, SLICES), block = 256 (8 waves). Each wave owns one 16-row
//   tile; A fragments stay register-resident across the column loop.
//   Mining is done on squared distances (sqrt is monotonic).
// ---------------------------------------------------------------------------
__global__ void __launch_bounds__(256)
triplet_mine_kernel(const unsigned short* __restrict__ Fhi,
                    const unsigned short* __restrict__ Flo,
                    const float* __restrict__ sqn,
                    const int*   __restrict__ labels,
                    float* __restrict__ posPart,
                    float* __restrict__ negPart) {
    const int lane  = threadIdx.x & 31;
    const int wave  = threadIdx.x >> 5;
    const int hi    = (lane >> 4) & 1;    // which 16-lane half
    const int ln    = lane & 15;
    const int itile = blockIdx.x * 8 + wave;   // 0..255
    const int ibase = itile * 16;
    const int slice = blockIdx.y;              // 0..SLICES-1
    const int jt0   = slice * JT_PER_SLICE;

    // ---- A fragments: rows ibase..ibase+15, all of K=0..255, hi+lo parts.
    // 16-bit A layout: lanes 0-15 hold M=lane, K = {kb..kb+7, kb+16..kb+23};
    // lanes 16-31 hold M=lane-16, K = {kb+8..kb+15, kb+24..kb+31}.
    const int arow = ibase + ln;
    const unsigned short* ahp = Fhi + arow * FEAT;
    const unsigned short* alp = Flo + arow * FEAT;

    FragU ah[KCHUNKS], al[KCHUNKS];
#pragma unroll
    for (int ch = 0; ch < KCHUNKS; ++ch) {
        const int kb = ch * 32;
        const int o0 = kb + hi * 8;
        const int o1 = kb + 16 + hi * 8;
        ah[ch].q[0] = *(const uint4*)(ahp + o0);
        ah[ch].q[1] = *(const uint4*)(ahp + o1);
        al[ch].q[0] = *(const uint4*)(alp + o0);
        al[ch].q[1] = *(const uint4*)(alp + o1);
    }

    // C layout: VGPR r, lanes 0-15 -> (M=r, N=lane); lanes 16-31 -> (M=r+8).
    float rn[8]; int rlab[8]; int mrow[8];
#pragma unroll
    for (int r = 0; r < 8; ++r) {
        mrow[r] = ibase + r + hi * 8;
        rn[r]   = sqn[mrow[r]];
        rlab[r] = labels[mrow[r]];
    }

    float pmax[8], nmin[8];
#pragma unroll
    for (int r = 0; r < 8; ++r) { pmax[r] = -1e30f; nmin[r] = 1e30f; }

    for (int jt = jt0; jt < jt0 + JT_PER_SLICE; ++jt) {
        const int jbase = jt * 16;
        const int jcol  = jbase + ln;         // this lane's output column
        // B (K x 16) column n = row (jbase+n) of F: 16-bit B layout matches A:
        // K = {kb+8h..kb+8h+7, kb+16+8h..kb+23+8h} per lane half h
        const unsigned short* bhp = Fhi + jcol * FEAT;
        const unsigned short* blp = Flo + jcol * FEAT;
        const float cn  = sqn[jcol];
        const int  clab = labels[jcol];

        v8f c = {};
#pragma unroll
        for (int ch = 0; ch < KCHUNKS; ++ch) {
            const int o = ch * 32 + hi * 8;
            FragU bh, bl;
            bh.q[0] = *(const uint4*)(bhp + o);
            bh.q[1] = *(const uint4*)(bhp + o + 16);
            bl.q[0] = *(const uint4*)(blp + o);
            bl.q[1] = *(const uint4*)(blp + o + 16);
            // split-precision: a.b ~= ah.bh + ah.bl + al.bh  (drop al.bl)
            c = __builtin_amdgcn_wmma_f32_16x16x32_bf16(false, ah[ch].v, false, bh.v, (short)0, c, false, false);
            c = __builtin_amdgcn_wmma_f32_16x16x32_bf16(false, ah[ch].v, false, bl.v, (short)0, c, false, false);
            c = __builtin_amdgcn_wmma_f32_16x16x32_bf16(false, al[ch].v, false, bh.v, (short)0, c, false, false);
        }

        // Branchless epilogue (keeps EXEC all-1s between WMMAs).
#pragma unroll
        for (int r = 0; r < 8; ++r) {
            float sq = fmaxf(rn[r] + cn - 2.0f * c[r], 1e-12f);
            const bool same = (rlab[r] == clab);
            const bool diag = (mrow[r] == jcol);
            const float pc = (same && !diag) ? sq : -1e30f;
            const float nc = same ? 1e30f : sq;
            pmax[r] = fmaxf(pmax[r], pc);
            nmin[r] = fminf(nmin[r], nc);
        }
    }

    // Reduce across the 16 columns held by this lane half.
#pragma unroll
    for (int r = 0; r < 8; ++r) {
#pragma unroll
        for (int off = 8; off >= 1; off >>= 1) {
            pmax[r] = fmaxf(pmax[r], __shfl_xor(pmax[r], off, 32));
            nmin[r] = fminf(nmin[r], __shfl_xor(nmin[r], off, 32));
        }
    }

    if (ln == 0) {   // lanes 0 (rows r) and 16 (rows r+8) write
#pragma unroll
        for (int r = 0; r < 8; ++r) {
            posPart[slice * BATCH + mrow[r]] = pmax[r];
            negPart[slice * BATCH + mrow[r]] = nmin[r];
        }
    }
}

// ---------------------------------------------------------------------------
// Kernel 2: combine slice partials, compute masked mean loss.
// ---------------------------------------------------------------------------
__global__ void __launch_bounds__(1024)
triplet_finalize_kernel(const float* __restrict__ posPart,
                        const float* __restrict__ negPart,
                        float* __restrict__ out) {
    const int tid = threadIdx.x;
    float total = 0.0f, cnt = 0.0f;
    for (int row = tid; row < BATCH; row += 1024) {
        float hp = -1e30f, hn = 1e30f;
#pragma unroll
        for (int s = 0; s < SLICES; ++s) {
            hp = fmaxf(hp, posPart[s * BATCH + row]);
            hn = fminf(hn, negPart[s * BATCH + row]);
        }
        if (hp > -1e29f && hn < 1e29f) {
            float l = sqrtf(fmaxf(hp, 1e-12f)) - sqrtf(fmaxf(hn, 1e-12f)) + MARGIN;
            total += fmaxf(l, 0.0f);
            cnt   += 1.0f;
        }
    }
    __shared__ float st[1024];
    __shared__ float sc[1024];
    st[tid] = total; sc[tid] = cnt;
    __syncthreads();
    for (int off = 512; off > 0; off >>= 1) {
        if (tid < off) { st[tid] += st[tid + off]; sc[tid] += sc[tid + off]; }
        __syncthreads();
    }
    if (tid == 0) out[0] = (sc[0] > 0.0f) ? (st[0] / sc[0]) : 0.0f;
}

// ---------------------------------------------------------------------------
extern "C" void kernel_launch(void* const* d_in, const int* in_sizes, int n_in,
                              void* d_out, int out_size, void* d_ws, size_t ws_size,
                              hipStream_t stream) {
    (void)in_sizes; (void)n_in; (void)out_size; (void)ws_size;
    const float* feats  = (const float*)d_in[0];
    const int*   labels = (const int*)d_in[1];
    float*       out    = (float*)d_out;

    // Workspace layout (~4.3 MB total):
    char* ws = (char*)d_ws;
    unsigned short* Fhi = (unsigned short*)(ws);                       // 2 MB
    unsigned short* Flo = (unsigned short*)(ws + 2097152);             // 2 MB
    float* sqn     = (float*)(ws + 4194304);                           // 16 KB
    float* posPart = (float*)(ws + 4194304 + 16384);                   // 128 KB
    float* negPart = (float*)(ws + 4194304 + 16384 + 131072);          // 128 KB

    triplet_prep_kernel<<<BATCH, 256, 0, stream>>>(feats, Fhi, Flo, sqn);
    triplet_mine_kernel<<<dim3(32, SLICES), 256, 0, stream>>>(
        Fhi, Flo, sqn, labels, posPart, negPart);
    triplet_finalize_kernel<<<1, 1024, 0, stream>>>(posPart, negPart, out);
}
